// AttnBlock_46213848105821
// MI455X (gfx1250) — matrix operand order repaired and checked by reading, not compiled, not run on hardware
//
#include <hip/hip_runtime.h>
#include <hip/hip_bf16.h>

// ---------------------------------------------------------------------------
// AttnBlock for MI455X (gfx1250): all GEMMs on v_wmma_f32_16x16x32_bf16.
// b=4, c=512, n=64*64=4096, 32 groups. Double-buffered 128x128x32 tiles,
// 8 waves/block (wave32), packed bf16 staging (b64 loads + ds_store_b32).
// ---------------------------------------------------------------------------

typedef __bf16 bf16;
typedef __attribute__((ext_vector_type(16))) __bf16 v16bf;
typedef __attribute__((ext_vector_type(8)))  __bf16 v8bf;
typedef __attribute__((ext_vector_type(2)))  __bf16 v2bf;
typedef __attribute__((ext_vector_type(8)))  float  v8f;

#define BATCH   4
#define CCH     512
#define NPIX    4096
#define GROUPS  32

#define BM      128
#define BN      128
#define BK      32
#define LDSK    40          // padded LDS row stride (halves): conflict-free b128 frag reads
#define THREADS 256         // 8 waves: 2 M-waves x 4 N-waves

#define WMMA_BF16(ACC, A, B) \
  (ACC) = __builtin_amdgcn_wmma_f32_16x16x32_bf16(false, (A), false, (B), (short)0, (ACC), false, false)

__device__ __forceinline__ unsigned pack2bf(float a, float b) {
  union { v2bf v; unsigned u; } p;
  p.v[0] = (bf16)a; p.v[1] = (bf16)b;
  return p.u;                          // lowers to v_cvt_pk_bf16_f32
}
__device__ __forceinline__ unsigned pack2h(bf16 a, bf16 b) {
  union { v2bf v; unsigned u; } p;
  p.v[0] = a; p.v[1] = b;
  return p.u;
}

// ------------------------- register-staged tiles ---------------------------
// bf16 tile [128 x BK] from row-major bf16 source (row stride `stride`).
struct TileB16 { uint4 r[2]; };
__device__ __forceinline__ TileB16 load_tile_bf16(const bf16* src, size_t stride,
                                                  size_t row0, int k0) {
  const int kq = threadIdx.x & 3;     // 8-half chunk
  const int r  = threadIdx.x >> 2;    // 0..63
  TileB16 t;
  t.r[0] = *(const uint4*)(src + (row0 + r)      * stride + k0 + kq * 8);
  t.r[1] = *(const uint4*)(src + (row0 + r + 64) * stride + k0 + kq * 8);
  return t;
}
__device__ __forceinline__ void store_tile_bf16(bf16* lds, TileB16 t) {
  const int kq = threadIdx.x & 3;
  const int r  = threadIdx.x >> 2;
  *(uint4*)(lds + r        * LDSK + kq * 8) = t.r[0];
  *(uint4*)(lds + (r + 64) * LDSK + kq * 8) = t.r[1];
}

// fp32 weight tile [BM x BK] (row-major, stride Ktot) -> packed bf16 pairs.
struct TileW { unsigned u[8]; };
__device__ __forceinline__ TileW load_tile_w(const float* W, int Ktot, int m0, int k0) {
  const int kk = (threadIdx.x & 15) * 2;   // even k within BK
  const int mr = threadIdx.x >> 4;         // 0..15
  TileW t;
#pragma unroll
  for (int i = 0; i < 8; ++i) {
    const float2 f = *(const float2*)(W + (size_t)(m0 + mr + 16 * i) * Ktot + k0 + kk);
    t.u[i] = pack2bf(f.x, f.y);
  }
  return t;
}
__device__ __forceinline__ void store_tile_w(bf16* lds, TileW t) {
  const int kk = (threadIdx.x & 15) * 2;
  const int mr = threadIdx.x >> 4;
#pragma unroll
  for (int i = 0; i < 8; ++i)
    *(unsigned*)(lds + (mr + 16 * i) * LDSK + kk) = t.u[i];
}

// GN-fused transpose tile: x[k][n] fp32 -> LDS [n][k] bf16 pairs.
struct TileGN { unsigned u[8]; };
__device__ __forceinline__ TileGN load_tile_gn(const float* x, const float* stats,
                                               const float* gamma, const float* beta,
                                               int b, int k0, int n0) {
  const int nn = threadIdx.x & 127;
  const int kh = (threadIdx.x >> 7) * 16;
  TileGN t;
#pragma unroll
  for (int p = 0; p < 8; ++p) {
    const int kk = kh + 2 * p;
    const int c  = k0 + kk;                 // c even: c, c+1 share a group of 16
    const float mu = stats[(b * GROUPS + (c >> 4)) * 2 + 0];
    const float rs = stats[(b * GROUPS + (c >> 4)) * 2 + 1];
    const float x0 = x[((size_t)b * CCH + c)     * NPIX + n0 + nn];
    const float x1 = x[((size_t)b * CCH + c + 1) * NPIX + n0 + nn];
    t.u[p] = pack2bf((x0 - mu) * rs * gamma[c] + beta[c],
                     (x1 - mu) * rs * gamma[c + 1] + beta[c + 1]);
  }
  return t;
}
// bf16 transpose tile: o[k][n] bf16 -> LDS [n][k] pairs.
struct TileT16 { unsigned u[8]; };
__device__ __forceinline__ TileT16 load_tile_t16(const bf16* src, int k0, int n0) {
  const int nn = threadIdx.x & 127;
  const int kh = (threadIdx.x >> 7) * 16;
  TileT16 t;
#pragma unroll
  for (int p = 0; p < 8; ++p) {
    const int kk = kh + 2 * p;
    t.u[p] = pack2h(src[(size_t)(k0 + kk)     * NPIX + n0 + nn],
                    src[(size_t)(k0 + kk + 1) * NPIX + n0 + nn]);
  }
  return t;
}
__device__ __forceinline__ void store_tile_pairs(bf16* lds, const unsigned* u) {
  const int nn = threadIdx.x & 127;
  const int kh = (threadIdx.x >> 7) * 16;
#pragma unroll
  for (int p = 0; p < 8; ++p)
    *(unsigned*)(lds + nn * LDSK + kh + 2 * p) = u[p];
}

// ------------------------- WMMA fragment loads -----------------------------
// 16-bit A 16x32: lane L row M=L&15; lo 8 halves K[8*khi..+7], hi K[16+8*khi..+7].
__device__ __forceinline__ v16bf load_frag_a(const bf16* lds, int mbase) {
  const int lane = threadIdx.x & 31;
  const int m    = mbase + (lane & 15);
  const int khi  = (lane >> 4) * 8;
  v8bf lo = *(const v8bf*)(lds + m * LDSK + khi);
  v8bf hi = *(const v8bf*)(lds + m * LDSK + 16 + khi);
  v16bf f;
#pragma unroll
  for (int i = 0; i < 8; ++i) { f[i] = lo[i]; f[8 + i] = hi[i]; }
  return f;
}
// 16-bit B 32x16: lane L col N=L&15; lo 8 halves K[8*kh..+7], hi K[16+8*kh..+7].
__device__ __forceinline__ v16bf load_frag_b(const bf16* lds, int nbase) {
  const int lane = threadIdx.x & 31;
  const int n    = nbase + (lane & 15);
  const int kh   = (lane >> 4) * 8;
  v8bf lo = *(const v8bf*)(lds + n * LDSK + kh);
  v8bf hi = *(const v8bf*)(lds + n * LDSK + 16 + kh);
  v16bf f;
#pragma unroll
  for (int i = 0; i < 8; ++i) { f[i] = lo[i]; f[8 + i] = hi[i]; }
  return f;
}

// One k-step of the 64x32-per-wave WMMA block.
__device__ __forceinline__ void wmma_step(const bf16* lA, const bf16* lB,
                                          int wm, int wn, v8f (&acc)[4][2]) {
  v16bf bfr[2];
#pragma unroll
  for (int ni = 0; ni < 2; ++ni) bfr[ni] = load_frag_b(lB, wn + ni * 16);
#pragma unroll
  for (int mi = 0; mi < 4; ++mi) {
    v16bf af = load_frag_a(lA, wm + mi * 16);
#pragma unroll
    for (int ni = 0; ni < 2; ++ni) WMMA_BF16(acc[mi][ni], af, bfr[ni]);
  }
}

#define ACC_DECL_INIT                       \
  v8f acc[4][2];                            \
  _Pragma("unroll")                         \
  for (int mi = 0; mi < 4; ++mi)            \
    _Pragma("unroll")                       \
    for (int ni = 0; ni < 2; ++ni)          \
      _Pragma("unroll")                     \
      for (int g = 0; g < 8; ++g) acc[mi][ni][g] = 0.f;

// ------------------------- Kernel 1: GroupNorm stats -----------------------
__global__ __launch_bounds__(256)
void gn_stats_kernel(const float* __restrict__ x, float* __restrict__ stats) {
  const int bg = blockIdx.x;               // 0..127
  const int b  = bg >> 5, g = bg & 31;
  const float* base = x + ((size_t)b * CCH + g * 16) * NPIX;  // contiguous 65536
  const int total = 16 * NPIX;
  float s = 0.f, s2 = 0.f;
  const float4* b4 = (const float4*)base;
  for (int i = threadIdx.x; i < total / 4; i += 256) {
    float4 v = b4[i];
    s  += v.x + v.y + v.z + v.w;
    s2 += v.x * v.x + v.y * v.y + v.z * v.z + v.w * v.w;
  }
  __shared__ float rs[256], rq[256];
  const int t = threadIdx.x;
  rs[t] = s; rq[t] = s2;
  __syncthreads();
  for (int off = 128; off; off >>= 1) {
    if (t < off) { rs[t] += rs[t + off]; rq[t] += rq[t + off]; }
    __syncthreads();
  }
  if (t == 0) {
    float mean = rs[0] / total;
    float var  = rq[0] / total - mean * mean;
    stats[bg * 2 + 0] = mean;
    stats[bg * 2 + 1] = rsqrtf(var + 1e-6f);
  }
}

// ------------------------- Kernel 2: fused GN + QKV GEMM -------------------
__global__ __launch_bounds__(THREADS)
void qkv_kernel(const float* __restrict__ x, const float* __restrict__ stats,
                const float* __restrict__ gamma, const float* __restrict__ beta,
                const float* __restrict__ wq, const float* __restrict__ bq,
                const float* __restrict__ wk, const float* __restrict__ bk,
                const float* __restrict__ wv, const float* __restrict__ bv,
                bf16* __restrict__ qT, bf16* __restrict__ kT, bf16* __restrict__ vN) {
  const int bz = blockIdx.z;
  const int b = bz / 3, proj = bz % 3;
  const float* W    = (proj == 0) ? wq : (proj == 1) ? wk : wv;
  const float* bias = (proj == 0) ? bq : (proj == 1) ? bk : bv;
  const int n0 = blockIdx.x * BN;
  const int m0 = blockIdx.y * BM;

  __shared__ __align__(16) bf16 lA[2][BM * LDSK];
  __shared__ __align__(16) bf16 lB[2][BN * LDSK];

  const int lane = threadIdx.x & 31;
  const int wave = threadIdx.x >> 5;
  const int wm = (wave >> 2) * 64, wn = (wave & 3) * 32;
  ACC_DECL_INIT;

  TileW  wa = load_tile_w(W, CCH, m0, 0);
  TileGN gb = load_tile_gn(x, stats, gamma, beta, b, 0, n0);
  store_tile_w(lA[0], wa);
  store_tile_pairs(lB[0], gb.u);
  int cur = 0;
  for (int k0 = 0; k0 < CCH; k0 += BK) {
    const bool nxt = (k0 + BK) < CCH;
    if (nxt) {
      wa = load_tile_w(W, CCH, m0, k0 + BK);
      gb = load_tile_gn(x, stats, gamma, beta, b, k0 + BK, n0);
    }
    __syncthreads();
    wmma_step(lA[cur], lB[cur], wm, wn, acc);
    if (nxt) {
      store_tile_w(lA[cur ^ 1], wa);
      store_tile_pairs(lB[cur ^ 1], gb.u);
      cur ^= 1;
    }
  }

  // Epilogue (+bias). D layout: row m = base + g + 8*(lane>>4), col n = base + (lane&15).
#pragma unroll
  for (int mi = 0; mi < 4; ++mi) {
#pragma unroll
    for (int ni = 0; ni < 2; ++ni) {
      const int n     = n0 + wn + ni * 16 + (lane & 15);
      const int mbase = m0 + wm + mi * 16 + 8 * (lane >> 4);
      if (proj <= 1) {                       // transposed store [n][c]: 8 contiguous bf16
        bf16* dst = ((proj == 0) ? qT : kT) + ((size_t)b * NPIX + n) * CCH + mbase;
        v8bf pk;
#pragma unroll
        for (int g = 0; g < 8; ++g) pk[g] = (bf16)(acc[mi][ni][g] + bias[mbase + g]);
        *(v8bf*)dst = pk;
      } else {                               // natural store [c][n]
#pragma unroll
        for (int g = 0; g < 8; ++g) {
          const int m = mbase + g;
          vN[((size_t)b * CCH + m) * NPIX + n] = (bf16)(acc[mi][ni][g] + bias[m]);
        }
      }
    }
  }
}

// ------------------------- Kernel 3: sim = scale * Q^T K -------------------
__global__ __launch_bounds__(THREADS)
void sim_kernel(const bf16* __restrict__ qT, const bf16* __restrict__ kT,
                bf16* __restrict__ attn) {
  const int b  = blockIdx.z;
  const int r0 = blockIdx.x * BN;            // key block
  const int p0 = blockIdx.y * BM;            // query block
  __shared__ __align__(16) bf16 lA[2][BM * LDSK];
  __shared__ __align__(16) bf16 lB[2][BN * LDSK];
  const int lane = threadIdx.x & 31;
  const int wave = threadIdx.x >> 5;
  const int wm = (wave >> 2) * 64, wn = (wave & 3) * 32;
  ACC_DECL_INIT;

  const bf16* qb = qT + (size_t)b * NPIX * CCH;
  const bf16* kb = kT + (size_t)b * NPIX * CCH;
  TileB16 ta = load_tile_bf16(qb, CCH, p0, 0);
  TileB16 tb = load_tile_bf16(kb, CCH, r0, 0);
  store_tile_bf16(lA[0], ta);
  store_tile_bf16(lB[0], tb);
  int cur = 0;
  for (int k0 = 0; k0 < CCH; k0 += BK) {
    const bool nxt = (k0 + BK) < CCH;
    if (nxt) {
      ta = load_tile_bf16(qb, CCH, p0, k0 + BK);
      tb = load_tile_bf16(kb, CCH, r0, k0 + BK);
    }
    __syncthreads();
    wmma_step(lA[cur], lB[cur], wm, wn, acc);
    if (nxt) {
      store_tile_bf16(lA[cur ^ 1], ta);
      store_tile_bf16(lB[cur ^ 1], tb);
      cur ^= 1;
    }
  }

  const float scale = 0.044194173824159216f;  // 512^-0.5
#pragma unroll
  for (int mi = 0; mi < 4; ++mi)
#pragma unroll
    for (int ni = 0; ni < 2; ++ni) {
      const int r     = r0 + wn + ni * 16 + (lane & 15);
      const int pbase = p0 + wm + mi * 16 + 8 * (lane >> 4);
#pragma unroll
      for (int g = 0; g < 8; ++g)
        attn[((size_t)b * NPIX + (pbase + g)) * NPIX + r] = (bf16)(acc[mi][ni][g] * scale);
    }
}

// ------------------------- Kernel 4: row softmax in place ------------------
__global__ __launch_bounds__(256)
void softmax_kernel(bf16* __restrict__ attn) {
  const size_t row = blockIdx.x;             // 0 .. BATCH*NPIX-1
  v8bf* p8 = (v8bf*)(attn + row * NPIX);
  const int t = threadIdx.x;
  float vals[16];
  float mx = -1e30f;
#pragma unroll
  for (int i = 0; i < 2; ++i) {
    v8bf ch = p8[t + 256 * i];
#pragma unroll
    for (int j = 0; j < 8; ++j) {
      float f = (float)ch[j];
      vals[i * 8 + j] = f;
      mx = fmaxf(mx, f);
    }
  }
  __shared__ float red[256];
  red[t] = mx; __syncthreads();
  for (int off = 128; off; off >>= 1) {
    if (t < off) red[t] = fmaxf(red[t], red[t + off]);
    __syncthreads();
  }
  mx = red[0]; __syncthreads();
  float sum = 0.f;
#pragma unroll
  for (int i = 0; i < 16; ++i) { float e = __expf(vals[i] - mx); vals[i] = e; sum += e; }
  red[t] = sum; __syncthreads();
  for (int off = 128; off; off >>= 1) {
    if (t < off) red[t] += red[t + off];
    __syncthreads();
  }
  const float inv = 1.0f / red[0];
#pragma unroll
  for (int i = 0; i < 2; ++i) {
    v8bf ch;
#pragma unroll
    for (int j = 0; j < 8; ++j) ch[j] = (bf16)(vals[i * 8 + j] * inv);
    p8[t + 256 * i] = ch;
  }
}

// ------------------------- Kernel 5: O = attn * V^T ------------------------
__global__ __launch_bounds__(THREADS)
void av_kernel(const bf16* __restrict__ attn, const bf16* __restrict__ vN,
               bf16* __restrict__ oN) {
  const int b  = blockIdx.z;
  const int c0 = blockIdx.x * BN;            // channel block (512/128 = 4)
  const int p0 = blockIdx.y * BM;            // query block
  __shared__ __align__(16) bf16 lA[2][BM * LDSK];
  __shared__ __align__(16) bf16 lB[2][BN * LDSK];
  const int lane = threadIdx.x & 31;
  const int wave = threadIdx.x >> 5;
  const int wm = (wave >> 2) * 64, wn = (wave & 3) * 32;
  ACC_DECL_INIT;

  const bf16* ab = attn + (size_t)b * NPIX * NPIX;
  const bf16* vb = vN + (size_t)b * CCH * NPIX;
  TileB16 ta = load_tile_bf16(ab, NPIX, p0, 0);
  TileB16 tb = load_tile_bf16(vb, NPIX, c0, 0);
  store_tile_bf16(lA[0], ta);
  store_tile_bf16(lB[0], tb);
  int cur = 0;
  for (int k0 = 0; k0 < NPIX; k0 += BK) {    // 128 k-steps over keys
    const bool nxt = (k0 + BK) < NPIX;
    if (nxt) {
      ta = load_tile_bf16(ab, NPIX, p0, k0 + BK);
      tb = load_tile_bf16(vb, NPIX, c0, k0 + BK);
    }
    __syncthreads();
    wmma_step(lA[cur], lB[cur], wm, wn, acc);
    if (nxt) {
      store_tile_bf16(lA[cur ^ 1], ta);
      store_tile_bf16(lB[cur ^ 1], tb);
      cur ^= 1;
    }
  }

#pragma unroll
  for (int mi = 0; mi < 4; ++mi)
#pragma unroll
    for (int ni = 0; ni < 2; ++ni) {
      const int cch   = c0 + wn + ni * 16 + (lane & 15);
      const int pbase = p0 + wm + mi * 16 + 8 * (lane >> 4);
      bf16* dst = oN + ((size_t)b * CCH + cch) * NPIX + pbase;  // 8 contiguous p
      v8bf pk;
#pragma unroll
      for (int g = 0; g < 8; ++g) pk[g] = (bf16)acc[mi][ni][g];
      *(v8bf*)dst = pk;
    }
}

// ------------------------- Kernel 6: out = x + Wo*O + bo -------------------
__global__ __launch_bounds__(THREADS)
void out_proj_kernel(const bf16* __restrict__ oN, const float* __restrict__ wo,
                     const float* __restrict__ bo, const float* __restrict__ x,
                     float* __restrict__ out) {
  const int b  = blockIdx.z;
  const int n0 = blockIdx.x * BN;
  const int m0 = blockIdx.y * BM;
  __shared__ __align__(16) bf16 lA[2][BM * LDSK];
  __shared__ __align__(16) bf16 lB[2][BN * LDSK];
  const int lane = threadIdx.x & 31;
  const int wave = threadIdx.x >> 5;
  const int wm = (wave >> 2) * 64, wn = (wave & 3) * 32;
  ACC_DECL_INIT;

  const bf16* ob = oN + (size_t)b * CCH * NPIX;
  TileW   wa = load_tile_w(wo, CCH, m0, 0);
  TileT16 tb = load_tile_t16(ob, 0, n0);
  store_tile_w(lA[0], wa);
  store_tile_pairs(lB[0], tb.u);
  int cur = 0;
  for (int k0 = 0; k0 < CCH; k0 += BK) {
    const bool nxt = (k0 + BK) < CCH;
    if (nxt) {
      wa = load_tile_w(wo, CCH, m0, k0 + BK);
      tb = load_tile_t16(ob, k0 + BK, n0);
    }
    __syncthreads();
    wmma_step(lA[cur], lB[cur], wm, wn, acc);
    if (nxt) {
      store_tile_w(lA[cur ^ 1], wa);
      store_tile_pairs(lB[cur ^ 1], tb.u);
      cur ^= 1;
    }
  }

#pragma unroll
  for (int mi = 0; mi < 4; ++mi)
#pragma unroll
    for (int ni = 0; ni < 2; ++ni) {
      const int n     = n0 + wn + ni * 16 + (lane & 15);
      const int mbase = m0 + wm + mi * 16 + 8 * (lane >> 4);
#pragma unroll
      for (int g = 0; g < 8; ++g) {
        const int m = mbase + g;
        const size_t idx = ((size_t)b * CCH + m) * NPIX + n;
        out[idx] = acc[mi][ni][g] + bo[m] + x[idx];   // residual
      }
    }
}

// ------------------------- launcher ----------------------------------------
extern "C" void kernel_launch(void* const* d_in, const int* in_sizes, int n_in,
                              void* d_out, int out_size, void* d_ws, size_t ws_size,
                              hipStream_t stream) {
  (void)in_sizes; (void)n_in; (void)out_size; (void)ws_size;
  const float* x     = (const float*)d_in[0];
  const float* gamma = (const float*)d_in[1];
  const float* beta  = (const float*)d_in[2];
  const float* wq = (const float*)d_in[3];  const float* bq = (const float*)d_in[4];
  const float* wk = (const float*)d_in[5];  const float* bk = (const float*)d_in[6];
  const float* wv = (const float*)d_in[7];  const float* bv = (const float*)d_in[8];
  const float* wo = (const float*)d_in[9];  const float* bo = (const float*)d_in[10];
  float* out = (float*)d_out;

  // Workspace carve-up (~201 MB total; assumes ws_size covers it):
  //   stats[256 f32] | qT | kT | v | o (each B*N*C bf16 = 16.8 MB) | attn (B*N*N bf16 = 134 MB)
  char* ws = (char*)d_ws;
  float* stats = (float*)ws;
  const size_t sz = (size_t)BATCH * NPIX * CCH;
  bf16* qT   = (bf16*)(ws + 1024);
  bf16* kT   = qT + sz;
  bf16* vN   = kT + sz;
  bf16* oN   = vN + sz;
  bf16* attn = oN + sz;

  gn_stats_kernel<<<dim3(BATCH * GROUPS), 256, 0, stream>>>(x, stats);
  qkv_kernel<<<dim3(NPIX / BN, CCH / BM, BATCH * 3), THREADS, 0, stream>>>(
      x, stats, gamma, beta, wq, bq, wk, bk, wv, bv, qT, kT, vN);
  sim_kernel<<<dim3(NPIX / BN, NPIX / BM, BATCH), THREADS, 0, stream>>>(qT, kT, attn);
  softmax_kernel<<<dim3(BATCH * NPIX), 256, 0, stream>>>(attn);
  av_kernel<<<dim3(CCH / BN, NPIX / BM, BATCH), THREADS, 0, stream>>>(attn, vN, oN);
  out_proj_kernel<<<dim3(NPIX / BN, CCH / BM, BATCH), THREADS, 0, stream>>>(
      oN, wo, bo, x, out);
}
